// GraphQA_51573967290929
// MI455X (gfx1250) — matrix operand order repaired and checked by reading, not compiled, not run on hardware
//
#include <hip/hip_runtime.h>
#include <hip/hip_bf16.h>

typedef __attribute__((ext_vector_type(16))) __bf16 v16bf;
typedef __attribute__((ext_vector_type(8)))  __bf16 v8bf;
typedef __attribute__((ext_vector_type(8)))  float  v8f;

#define N_NODES 100000
#define N_EDGES 500000
#define N_GRAPHS 64

// ---------------------------------------------------------------------------
// Tiled GEMM: Out[M,N_OUT] = relu(A[M,D_IN] @ W[D_IN,N_OUT] + bias)
// A is a *virtual* matrix depending on MODE:
//   MODE 0: A = A0 (plain, row-major [M,D_IN])
//   MODE 1: A = concat(h[src[row]], e[row], u[batch[src[row]]])   (edge_fn)
//   MODE 2: A = concat(h[row],      msg[row], u[batch[row]])      (node_fn)
//   MODE 3: A = relu(A0[row] * A1[k] + A2[k])  (fused 1->32 edge encoder)
// 256 threads = 8 waves; 128-row tile; each wave: 16 rows x N_OUT cols via
// bf16 WMMA, f32 accumulate. Staging uses float4 global loads; LDS holds the
// bf16 A tile (128 x D_IN) and W^T (N_OUT x D_IN).
// ---------------------------------------------------------------------------
template<int D_IN, int N_OUT, int MODE>
__global__ __launch_bounds__(256)
void gemm_wmma(const float* __restrict__ A0,
               const float* __restrict__ A1,
               const float* __restrict__ A2,
               const int*   __restrict__ srcIdx,
               const int*   __restrict__ batch,
               const float* __restrict__ W,
               const float* __restrict__ bias,
               float* __restrict__ Out,
               int M)
{
    static_assert(D_IN % 32 == 0, "K must be a multiple of 32");
    constexpr int SA = D_IN + 8;   // bf16 elems per A row (8-elem aligned rows)
    constexpr int SW = D_IN + 8;   // bf16 elems per W^T row
    constexpr int NT = N_OUT / 16; // 16x16 column tiles per wave
    constexpr int K4 = D_IN / 4;   // float4 chunks per A row

    extern __shared__ char smem[];
    __bf16* sA   = (__bf16*)smem;                        // 128 * SA
    __bf16* sW   = (__bf16*)(smem + 128 * SA * 2);       // N_OUT * SW
    int*    sSrc = (int*)(smem + 128 * SA * 2 + N_OUT * SW * 2); // 128
    int*    sG   = sSrc + 128;                           // 128

    const int tid     = threadIdx.x;
    const int rowBase = blockIdx.x * 128;

    // ---- stage per-row gather indices --------------------------------------
    if (MODE == 1) {
        for (int r = tid; r < 128; r += 256) {
            int row = rowBase + r;
            int s = (row < M) ? srcIdx[row] : 0;
            sSrc[r] = s;
            if (D_IN > 192) sG[r] = batch[s];
        }
    } else if (MODE == 2) {
        if (D_IN > 192) {
            for (int r = tid; r < 128; r += 256) {
                int row = rowBase + r;
                sG[r] = (row < M) ? batch[row] : 0;
            }
        }
    }
    __syncthreads();

    // ---- stage A tile (fused gather/concat), float4 loads, bf16x4 stores ---
    for (int i4 = tid; i4 < 128 * K4; i4 += 256) {
        int r = i4 / K4;
        int k = (i4 - r * K4) * 4;
        int row = rowBase + r;
        float4 v = make_float4(0.f, 0.f, 0.f, 0.f);
        if (row < M) {
            if (MODE == 0) {
                v = *(const float4*)(A0 + row * D_IN + k);
            } else if (MODE == 3) {
                float a = A0[row];
                float t0 = a * A1[k + 0] + A2[k + 0];
                float t1 = a * A1[k + 1] + A2[k + 1];
                float t2 = a * A1[k + 2] + A2[k + 2];
                float t3 = a * A1[k + 3] + A2[k + 3];
                v = make_float4(t0 > 0.f ? t0 : 0.f, t1 > 0.f ? t1 : 0.f,
                                t2 > 0.f ? t2 : 0.f, t3 > 0.f ? t3 : 0.f);
            } else if (MODE == 1) {
                int s = sSrc[r];
                if (k < 128)        v = *(const float4*)(A0 + s * 128 + k);
                else if (k < 192)   v = *(const float4*)(A1 + row * 64 + (k - 128));
                else if (D_IN > 192) v = *(const float4*)(A2 + sG[r] * 32 + (k - 192));
            } else { // MODE 2
                if (k < 128)        v = *(const float4*)(A0 + row * 128 + k);
                else if (k < 192)   v = *(const float4*)(A1 + row * 64 + (k - 128));
                else if (D_IN > 192) v = *(const float4*)(A2 + sG[r] * 32 + (k - 192));
            }
        }
        union { __bf16 b[4]; unsigned long long u; } pk;
        pk.b[0] = (__bf16)v.x; pk.b[1] = (__bf16)v.y;
        pk.b[2] = (__bf16)v.z; pk.b[3] = (__bf16)v.w;
        *(unsigned long long*)(sA + r * SA + k) = pk.u;   // 8B-aligned ds_store_b64
    }
    // ---- stage W transposed (float4 along N), fp32 -> bf16 -----------------
    for (int i4 = tid; i4 < D_IN * (N_OUT / 4); i4 += 256) {
        int k = i4 / (N_OUT / 4);
        int n = (i4 - k * (N_OUT / 4)) * 4;
        float4 w = *(const float4*)(W + k * N_OUT + n);
        sW[(n + 0) * SW + k] = (__bf16)w.x;
        sW[(n + 1) * SW + k] = (__bf16)w.y;
        sW[(n + 2) * SW + k] = (__bf16)w.z;
        sW[(n + 3) * SW + k] = (__bf16)w.w;
    }
    __syncthreads();

    // ---- WMMA compute ------------------------------------------------------
    const int wave = tid >> 5;      // 0..7 -> 16-row strip
    const int lane = tid & 31;
    const int mrow = lane & 15;
    const int hi   = lane >> 4;     // half-wave selector

    v8f acc[NT] = {};
    const __bf16* aBase = sA + (wave * 16 + mrow) * SA;

    for (int kk = 0; kk < D_IN; kk += 32) {
        // A frag: lane<16 -> K {kk..kk+15}; lane>=16 -> +16
        v8bf alo = *(const v8bf*)(aBase + kk + hi * 16);
        v8bf ahi = *(const v8bf*)(aBase + kk + hi * 16 + 8);
        v16bf afrag = __builtin_shufflevector(alo, ahi,
            0,1,2,3,4,5,6,7,8,9,10,11,12,13,14,15);
#pragma unroll
        for (int t = 0; t < NT; t++) {
            const __bf16* bBase = sW + (t * 16 + mrow) * SW + kk + hi * 16;
            v8bf blo = *(const v8bf*)(bBase);
            v8bf bhi = *(const v8bf*)(bBase + 8);
            v16bf bfrag = __builtin_shufflevector(blo, bhi,
                0,1,2,3,4,5,6,7,8,9,10,11,12,13,14,15);
            acc[t] = __builtin_amdgcn_wmma_f32_16x16x32_bf16(
                false, afrag, false, bfrag, (short)0, acc[t], false, false);
        }
    }

    // ---- epilogue: bias + relu + store ------------------------------------
#pragma unroll
    for (int t = 0; t < NT; t++) {
        int col = t * 16 + mrow;
        float bv = bias[col];
#pragma unroll
        for (int v = 0; v < 8; v++) {
            int row = rowBase + wave * 16 + v + hi * 8;
            if (row < M) {
                float o = acc[t][v] + bv;
                Out[row * N_OUT + col] = o > 0.f ? o : 0.f;
            }
        }
    }
}

// ---------------------------------------------------------------------------
// edges -> nodes (msg, global atomics) and edges -> graphs (e_g, LDS partials)
// ---------------------------------------------------------------------------
__global__ __launch_bounds__(256)
void scatter_edges(const float* __restrict__ efeat,
                   const int*   __restrict__ eidx,   // [2, E]
                   const int*   __restrict__ batch,
                   float* __restrict__ msg,          // [N,64] pre-zeroed
                   float* __restrict__ gcat,         // e_g at cols [0,64)
                   int gstr, int E)
{
    __shared__ float lacc[N_GRAPHS * 64];
    int tid = threadIdx.x;
    for (int i = tid; i < N_GRAPHS * 64; i += 256) lacc[i] = 0.f;
    __syncthreads();

    int per = (E + gridDim.x - 1) / gridDim.x;
    int e0 = blockIdx.x * per;
    int e1 = e0 + per; if (e1 > E) e1 = E;

    for (int i = e0 * 64 + tid; i < e1 * 64; i += 256) {
        int e = i >> 6;
        int c = i & 63;
        float v = efeat[i];
        int d = eidx[E + e];                     // dst node
        atomicAdd(&msg[d * 64 + c], v);
        int g = batch[eidx[e]];                  // graph of src node
        atomicAdd(&lacc[g * 64 + c], v);         // ds_add_f32
    }
    __syncthreads();
    for (int i = tid; i < N_GRAPHS * 64; i += 256) {
        float v = lacc[i];
        if (v != 0.f) atomicAdd(&gcat[(i >> 6) * gstr + (i & 63)], v);
    }
}

// ---------------------------------------------------------------------------
// nodes -> graphs (x_g): batch is sorted, so run-length accumulate in regs
// ---------------------------------------------------------------------------
__global__ __launch_bounds__(256)
void reduce_nodes(const float* __restrict__ h,
                  const int*   __restrict__ batch,
                  float* __restrict__ gcat,         // x_g at cols [64,192)
                  int gstr, int N)
{
    int col  = threadIdx.x & 127;
    int half = threadIdx.x >> 7;
    int per = (N + gridDim.x - 1) / gridDim.x;
    int n0 = blockIdx.x * per + half;
    int n1 = blockIdx.x * per + per; if (n1 > N) n1 = N;

    float sum = 0.f;
    int g = -1;
    for (int n = n0; n < n1; n += 2) {
        int gn = batch[n];
        if (gn != g) {
            if (g >= 0) atomicAdd(&gcat[g * gstr + 64 + col], sum);
            g = gn; sum = 0.f;
        }
        sum += h[n * 128 + col];
    }
    if (g >= 0) atomicAdd(&gcat[g * gstr + 64 + col], sum);
}

__global__ void copy_u(const float* __restrict__ u, float* __restrict__ gcat, int gstr)
{
    int i = blockIdx.x * 256 + threadIdx.x;     // 64*32 = 2048
    if (i < N_GRAPHS * 32)
        gcat[(i >> 5) * gstr + 192 + (i & 31)] = u[i];
}

__global__ void readout_k(const float* __restrict__ u,
                          const float* __restrict__ Wr,
                          const float* __restrict__ br,
                          float* __restrict__ out)
{
    int g = threadIdx.x;
    if (g < N_GRAPHS) {
        float s = br[0];
        for (int k = 0; k < 32; k++) s += u[g * 32 + k] * Wr[k];
        out[g] = s;
    }
}

// ---------------------------------------------------------------------------
static size_t gemm_smem(int d_in, int n_out)
{
    size_t sa = (size_t)d_in + 8;
    return 128 * sa * 2 + (size_t)n_out * sa * 2 + 1024;
}

extern "C" void kernel_launch(void* const* d_in, const int* in_sizes, int n_in,
                              void* d_out, int out_size, void* d_ws, size_t ws_size,
                              hipStream_t stream)
{
    (void)in_sizes; (void)n_in; (void)out_size; (void)ws_size;

    const float* x     = (const float*)d_in[0];
    const int*   eidx  = (const int*)d_in[1];     // [2, E]: src then dst
    const float* eattr = (const float*)d_in[2];   // [E, 1]
    const int*   batch = (const int*)d_in[3];     // [N]
    const float* Wn0 = (const float*)d_in[4];  const float* bn0 = (const float*)d_in[5];
    const float* Wn1 = (const float*)d_in[6];  const float* bn1 = (const float*)d_in[7];
    const float* We0 = (const float*)d_in[8];  const float* be0 = (const float*)d_in[9];
    const float* We1 = (const float*)d_in[10]; const float* be1 = (const float*)d_in[11];
    // mp layer l: 12 + 6*l + {We,be,Wnd,bnd,Wg,bg}
    const float* Wr = (const float*)d_in[36];  const float* br = (const float*)d_in[37];

    // workspace layout (floats)
    float* h    = (float*)d_ws;                       // [N,128]
    float* e    = h    + (size_t)N_NODES * 128;       // [E,64]
    float* msg  = e    + (size_t)N_EDGES * 64;        // [N,64] (also encoder tmp)
    float* gcat = msg  + (size_t)N_NODES * 64;        // [64, 224] max
    float* u    = gcat + (size_t)N_GRAPHS * 224;      // [64, 32]

    const int gbN = (N_NODES + 127) / 128;   // GEMM blocks over nodes
    const int gbE = (N_EDGES + 127) / 128;   // GEMM blocks over edges

    // ---- encoders ----------------------------------------------------------
    gemm_wmma<96, 64, 0><<<gbN, 256, gemm_smem(96, 64), stream>>>(
        x, nullptr, nullptr, nullptr, nullptr, Wn0, bn0, msg, N_NODES);
    gemm_wmma<64, 128, 0><<<gbN, 256, gemm_smem(64, 128), stream>>>(
        msg, nullptr, nullptr, nullptr, nullptr, Wn1, bn1, h, N_NODES);
    // edge encoder: layer1 (1->32) fused into A-staging of layer2 (32->64)
    gemm_wmma<32, 64, 3><<<gbE, 256, gemm_smem(32, 64), stream>>>(
        eattr, We0, be0, nullptr, nullptr, We1, be1, e, N_EDGES);

    // ---- message-passing layers -------------------------------------------
    for (int l = 0; l < 4; l++) {
        const float* We = (const float*)d_in[12 + 6 * l + 0];
        const float* be = (const float*)d_in[12 + 6 * l + 1];
        const float* Wn = (const float*)d_in[12 + 6 * l + 2];
        const float* bn = (const float*)d_in[12 + 6 * l + 3];
        const float* Wg = (const float*)d_in[12 + 6 * l + 4];
        const float* bg = (const float*)d_in[12 + 6 * l + 5];
        const int gstr = (l == 0) ? 192 : 224;

        // edge_fn: e = relu([h[src], e, u[bsrc]] @ We + be), in-place on e
        if (l == 0)
            gemm_wmma<192, 64, 1><<<gbE, 256, gemm_smem(192, 64), stream>>>(
                h, e, u, eidx, batch, We, be, e, N_EDGES);
        else
            gemm_wmma<224, 64, 1><<<gbE, 256, gemm_smem(224, 64), stream>>>(
                h, e, u, eidx, batch, We, be, e, N_EDGES);

        hipMemsetAsync(msg,  0, (size_t)N_NODES * 64 * sizeof(float), stream);
        hipMemsetAsync(gcat, 0, (size_t)N_GRAPHS * gstr * sizeof(float), stream);
        scatter_edges<<<128, 256, 0, stream>>>(e, eidx, batch, msg, gcat, gstr, N_EDGES);

        // node_fn: h = relu([h, msg, u[batch]] @ Wn + bn), in-place on h
        if (l == 0)
            gemm_wmma<192, 128, 2><<<gbN, 256, gemm_smem(192, 128), stream>>>(
                h, msg, u, nullptr, batch, Wn, bn, h, N_NODES);
        else
            gemm_wmma<224, 128, 2><<<gbN, 256, gemm_smem(224, 128), stream>>>(
                h, msg, u, nullptr, batch, Wn, bn, h, N_NODES);

        reduce_nodes<<<256, 256, 0, stream>>>(h, batch, gcat, gstr, N_NODES);
        if (l > 0) copy_u<<<8, 256, 0, stream>>>(u, gcat, gstr);

        // global_fn: u = relu([e_g, x_g, u] @ Wg + bg)
        if (l == 0)
            gemm_wmma<192, 32, 0><<<1, 256, gemm_smem(192, 32), stream>>>(
                gcat, nullptr, nullptr, nullptr, nullptr, Wg, bg, u, N_GRAPHS);
        else
            gemm_wmma<224, 32, 0><<<1, 256, gemm_smem(224, 32), stream>>>(
                gcat, nullptr, nullptr, nullptr, nullptr, Wg, bg, u, N_GRAPHS);
    }

    readout_k<<<1, 64, 0, stream>>>(u, Wr, br, (float*)d_out);
}
